// ImprovedClusteredCausalAttention_86071144612554
// MI455X (gfx1250) — matrix operand order repaired and checked by reading, not compiled, not run on hardware
//
#include <hip/hip_runtime.h>
#include <hip/hip_bf16.h>

// Problem constants (from reference)
#define NB    2
#define HH    8
#define LL    2048
#define EE    64
#define BPL   32      // hash planes
#define CC    256     // clusters
#define NH    16      // NB*HH
#define KITER 10
#define TOPKK 32

typedef __attribute__((ext_vector_type(16))) _Float16 v16h;
typedef __attribute__((ext_vector_type(8)))  _Float16 v8h;
typedef __attribute__((ext_vector_type(8)))  float    v8f;

// ---- WMMA helpers -----------------------------------------------------------
__device__ inline v16h frag_2x8(const _Float16* p0, const _Float16* p1) {
    v16h r;
    ((v8h*)&r)[0] = *(const v8h*)p0;   // halves 0..7
    ((v8h*)&r)[1] = *(const v8h*)p1;   // halves 8..15
    return r;
}
__device__ inline v16h frag_c16(const _Float16* p) { return frag_2x8(p, p + 8); }

__device__ inline v8f wmma16(v16h a, v16h b, v8f c) {
    return __builtin_amdgcn_wmma_f32_16x16x32_f16(false, a, false, b, (short)0, c,
                                                  false, false);
}

// ---- 1) hash bits: proj = q·plane + bias > 0, packed into uint32 ------------
__global__ void k_bits(const float* __restrict__ q, const float* __restrict__ planes,
                       unsigned* __restrict__ bits) {
    int idx = blockIdx.x * blockDim.x + threadIdx.x;       // (nh, l)
    if (idx >= NH * LL) return;
    int nh = idx / LL, l = idx % LL;
    int n = nh >> 3, h = nh & 7;
    const float* qr = q + ((size_t)(n * LL + l) * HH + h) * EE;
    float qv[EE];
#pragma unroll
    for (int e = 0; e < EE; ++e) qv[e] = qr[e];
    unsigned m = 0;
    for (int b = 0; b < BPL; ++b) {
        const float* pl = planes + b * (EE + 1);
        float acc = pl[EE];
#pragma unroll
        for (int e = 0; e < EE; ++e) acc += qv[e] * pl[e];
        if (acc > 0.0f) m |= (1u << b);
    }
    bits[idx] = m;
}

// ---- 2) transpose (N,L,H,E)->(N,H,L,E) and convert to f16 -------------------
__global__ void k_half(const float* __restrict__ q, const float* __restrict__ k,
                       const float* __restrict__ v, _Float16* __restrict__ qh,
                       _Float16* __restrict__ kh, _Float16* __restrict__ vh) {
    int idx = blockIdx.x * blockDim.x + threadIdx.x;       // ((nh*L)+l)*E+e
    if (idx >= NH * LL * EE) return;
    int e  = idx & (EE - 1);
    int l  = (idx >> 6) & (LL - 1);
    int nh = idx >> 17;
    int n = nh >> 3, h = nh & 7;
    size_t src = ((size_t)(n * LL + l) * HH + h) * EE + e;
    qh[idx] = (_Float16)q[src];
    kh[idx] = (_Float16)k[src];
    vh[idx] = (_Float16)v[src];
}

// ---- 3) k-means on binary codes (Hamming via popcount), one WG per (n,h) ----
__global__ void k_kmeans(const unsigned* __restrict__ bits, int* __restrict__ clusters,
                         int* __restrict__ counts_g) {
    int nh = blockIdx.x;
    int tid = threadIdx.x;                                  // 256 threads
    __shared__ unsigned cent[CC];
    __shared__ int      cnt[CC];
    __shared__ int      sums[CC][BPL];                      // per-bit tallies

    cent[tid] = bits[nh * LL + tid * (LL / CC)];            // init_idx = c*L//C
    __syncthreads();

    for (int it = 0; it < KITER; ++it) {
        cnt[tid] = 0;
#pragma unroll
        for (int j = 0; j < BPL; ++j) sums[tid][j] = 0;
        __syncthreads();
        for (int l = tid; l < LL; l += 256) {
            unsigned b = bits[nh * LL + l];
            int best = 0, bd = 0x7fffffff;
            for (int c = 0; c < CC; ++c) {
                unsigned ce = cent[c];
                int d = __popc(ce) - 2 * __popc(b & ce);
                if (d < bd) { bd = d; best = c; }
            }
            atomicAdd(&cnt[best], 1);
#pragma unroll
            for (int bit = 0; bit < BPL; ++bit)
                if ((b >> bit) & 1u) atomicAdd(&sums[best][bit], 1);
        }
        __syncthreads();
        int c = cnt[tid];
        if (c > 0) {
            unsigned nc = 0;
#pragma unroll
            for (int bit = 0; bit < BPL; ++bit)
                if (2 * sums[tid][bit] > c) nc |= (1u << bit);   // s > 0.5*cnt
            cent[tid] = nc;
        }
        __syncthreads();
    }
    // final assignment + global counts
    for (int l = tid; l < LL; l += 256) {
        unsigned b = bits[nh * LL + l];
        int best = 0, bd = 0x7fffffff;
        for (int c = 0; c < CC; ++c) {
            unsigned ce = cent[c];
            int d = __popc(ce) - 2 * __popc(b & ce);
            if (d < bd) { bd = d; best = c; }
        }
        clusters[nh * LL + l] = best;
        atomicAdd(&counts_g[nh * CC + best], 1);
    }
}

// ---- 4) qg accumulation (scatter-add), then normalize+convert ---------------
__global__ void k_qg_acc(const float* __restrict__ q, const int* __restrict__ clusters,
                         float* __restrict__ qg) {
    int idx = blockIdx.x * blockDim.x + threadIdx.x;       // (nh,l,e)
    if (idx >= NH * LL * EE) return;
    int e  = idx & (EE - 1);
    int l  = (idx >> 6) & (LL - 1);
    int nh = idx >> 17;
    int n = nh >> 3, h = nh & 7;
    int cl = clusters[nh * LL + l];
    float val = q[((size_t)(n * LL + l) * HH + h) * EE + e];
    atomicAdd(&qg[((size_t)nh * CC + cl) * EE + e], val);
}

__global__ void k_qg_norm(const float* __restrict__ qg, const int* __restrict__ counts,
                          _Float16* __restrict__ qgh) {
    int idx = blockIdx.x * blockDim.x + threadIdx.x;
    if (idx >= NH * CC * EE) return;
    float c = (float)counts[idx >> 6];
    qgh[idx] = (_Float16)(qg[idx] / fmaxf(c, 1.0f));
}

// ---- 5) QKc = qg(256x64) x Kh^T(64x2048) per (n,h) via WMMA -----------------
// one wave per 16x64 output strip: A fragments loaded once, reused over 4
// key tiles (8 chained WMMAs per wave)
__global__ void __launch_bounds__(32)
k_qkc(const _Float16* __restrict__ qgh, const _Float16* __restrict__ khh,
      float* __restrict__ QKc) {
    int lane = threadIdx.x;
    int lt4   = blockIdx.x;      // 32 groups of 4 key tiles
    int ctile = blockIdx.y;      // 16 cluster tiles
    int nh    = blockIdx.z;      // 16
    int col = lane & 15, g = lane >> 4;

    const _Float16* arow = qgh + ((size_t)nh * CC + ctile * 16 + col) * EE;
    v16h a0 = frag_2x8(arow + g * 8,      arow + 16 + g * 8);
    v16h a1 = frag_2x8(arow + 32 + g * 8, arow + 48 + g * 8);

#pragma unroll
    for (int s = 0; s < 4; ++s) {
        int ltile = lt4 * 4 + s;
        const _Float16* brow = khh + ((size_t)nh * LL + ltile * 16 + col) * EE;
        v16h b0 = frag_2x8(brow + g * 8,      brow + 16 + g * 8);
        v16h b1 = frag_2x8(brow + 32 + g * 8, brow + 48 + g * 8);
        v8f acc = {};
        acc = wmma16(a0, b0, acc);
        acc = wmma16(a1, b1, acc);
        float* orow = QKc + ((size_t)nh * CC + ctile * 16) * LL + ltile * 16 + col;
#pragma unroll
        for (int r = 0; r < 8; ++r)
            orow[(size_t)(r + g * 8) * LL] = acc[r];
    }
}

// ---- 6) top-32 per (n,h,c) row of 2048 scores -------------------------------
// row staged into LDS with CDNA5 async memory->LDS copies (ASYNCcnt tracked)
__global__ void k_topk(const float* __restrict__ QKc, int* __restrict__ topk) {
    int c = blockIdx.x, nh = blockIdx.y, tid = threadIdx.x; // 256 threads
    const float* row = QKc + ((size_t)nh * CC + c) * LL;
    __shared__ float sv[LL];
    __shared__ float rv[256];
    __shared__ int   ri[256];

    // async copy 8KB row -> LDS: 2 x b128 per thread, no VGPR round-trip
    {
        unsigned l0 = (unsigned)(uintptr_t)(&sv[0]) + tid * 16u;
        unsigned long long g0 = (unsigned long long)(row + tid * 4);
        asm volatile("global_load_async_to_lds_b128 %0, %1, off"
                     :: "v"(l0), "v"(g0) : "memory");
        unsigned l1 = l0 + 4096u;
        unsigned long long g1 = g0 + 4096ull;
        asm volatile("global_load_async_to_lds_b128 %0, %1, off"
                     :: "v"(l1), "v"(g1) : "memory");
        asm volatile("s_wait_asynccnt 0x0" ::: "memory");
    }
    __syncthreads();

    for (int kk = 0; kk < TOPKK; ++kk) {
        float bv = -3.0e38f; int bi = LL;
        for (int i = tid; i < LL; i += 256) {
            float val = sv[i];
            if (val > bv) { bv = val; bi = i; }
        }
        rv[tid] = bv; ri[tid] = bi;
        __syncthreads();
        for (int s = 128; s > 0; s >>= 1) {
            if (tid < s) {
                float v2 = rv[tid + s]; int i2 = ri[tid + s];
                if (v2 > rv[tid] || (v2 == rv[tid] && i2 < ri[tid])) {
                    rv[tid] = v2; ri[tid] = i2;
                }
            }
            __syncthreads();
        }
        if (tid == 0) { topk[((size_t)nh * CC + c) * TOPKK + kk] = ri[0]; sv[ri[0]] = -3.0e38f; }
        __syncthreads();
    }
}

// ---- 7) exclusive scan of counts -> offsets, init cursors -------------------
__global__ void k_scan(const int* __restrict__ counts, int* __restrict__ coff,
                       int* __restrict__ cursor) {
    int nh = blockIdx.x;
    if (threadIdx.x != 0) return;
    int off = 0;
    for (int c = 0; c < CC; ++c) {
        coff[nh * CC + c] = off;
        cursor[nh * CC + c] = off;
        off += counts[nh * CC + c];
    }
}

__global__ void k_scatter(const int* __restrict__ clusters, int* __restrict__ cursor,
                          int* __restrict__ tsorted) {
    int idx = blockIdx.x * blockDim.x + threadIdx.x;       // (nh,l)
    if (idx >= NH * LL) return;
    int nh = idx / LL, l = idx % LL;
    int cl = clusters[idx];
    int pos = atomicAdd(&cursor[nh * CC + cl], 1);
    tsorted[nh * LL + pos] = l;
}

// ---- 8) attention: one wave per (cluster, nh); shared top-32 key set --------
__global__ void __launch_bounds__(32)
k_attn(const _Float16* __restrict__ qhh, const _Float16* __restrict__ khh,
       const _Float16* __restrict__ vhh, const int* __restrict__ topk,
       const int* __restrict__ counts, const int* __restrict__ coff,
       const int* __restrict__ tsorted, float* __restrict__ out) {
    int c = blockIdx.x, nh = blockIdx.y;
    int cnt = counts[nh * CC + c];
    if (cnt == 0) return;
    int off = coff[nh * CC + c];
    int lane = threadIdx.x, col = lane & 15, g = lane >> 4;
    int n = nh >> 3, h = nh & 7;

    __shared__ int      s_tq[TOPKK];
    __shared__ int      s_tok[16];
    __shared__ _Float16 s_p[16 * 32];

    s_tq[lane] = topk[((size_t)nh * CC + c) * TOPKK + lane];
    __syncthreads();

    // loop-invariant B fragments: Ksel^T (64x32) and Vsel (32x64)
    v16h bk[2][2];
#pragma unroll
    for (int kk = 0; kk < 2; ++kk)
#pragma unroll
        for (int t = 0; t < 2; ++t) {
            int key = s_tq[t * 16 + col];
            bk[kk][t] = frag_2x8(khh + ((size_t)nh * LL + key) * EE + kk * 32 + g * 8, khh + ((size_t)nh * LL + key) * EE + kk * 32 + 16 + g * 8);
        }
    v16h bv[4];
#pragma unroll
    for (int t = 0; t < 4; ++t) {
        int e = t * 16 + col;
        v16h tmp;
#pragma unroll
        for (int i = 0; i < 16; ++i) {
            int key = s_tq[i < 8 ? g * 8 + i : 16 + g * 8 + (i - 8)];
            tmp[i] = vhh[((size_t)nh * LL + key) * EE + e];
        }
        bv[t] = tmp;
    }
    int tq0 = s_tq[col], tq1 = s_tq[16 + col];

    int ntiles = (cnt + 15) >> 4;
    for (int tile = 0; tile < ntiles; ++tile) {
        __syncthreads();
        int j = tile * 16 + col;
        int jj = j < cnt ? j : cnt - 1;
        int tok = tsorted[nh * LL + off + jj];
        if (lane < 16) s_tok[lane] = tok;
        __syncthreads();

        // QK = q_tile(16x64) x Ksel^T(64x32): two n-tiles, K chained over 64
        const _Float16* qrow = qhh + ((size_t)nh * LL + tok) * EE;
        v16h a0 = frag_2x8(qrow + g * 8,      qrow + 16 + g * 8);
        v16h a1 = frag_2x8(qrow + 32 + g * 8, qrow + 48 + g * 8);
        v8f acc0 = {}, acc1 = {};
        acc0 = wmma16(a0, bk[0][0], acc0);
        acc0 = wmma16(a1, bk[1][0], acc0);
        acc1 = wmma16(a0, bk[0][1], acc1);
        acc1 = wmma16(a1, bk[1][1], acc1);

        // causal mask + softmax(temp*QK) per row (rows live across 16 lanes)
        float p0[8], p1[8];
#pragma unroll
        for (int r = 0; r < 8; ++r) {
            int row = r + g * 8;
            int ltok = s_tok[row];
            float v0 = acc0[r]; if (tq0 > ltok) v0 = -10000000.0f;
            float v1 = acc1[r]; if (tq1 > ltok) v1 = -10000000.0f;
            v0 *= 0.125f; v1 *= 0.125f;
            float m = fmaxf(v0, v1);
            m = fmaxf(m, __shfl_xor(m, 1, 32));
            m = fmaxf(m, __shfl_xor(m, 2, 32));
            m = fmaxf(m, __shfl_xor(m, 4, 32));
            m = fmaxf(m, __shfl_xor(m, 8, 32));
            float e0 = __expf(v0 - m), e1 = __expf(v1 - m);
            float s = e0 + e1;
            s += __shfl_xor(s, 1, 32);
            s += __shfl_xor(s, 2, 32);
            s += __shfl_xor(s, 4, 32);
            s += __shfl_xor(s, 8, 32);
            p0[r] = e0 / s; p1[r] = e1 / s;
        }
        // re-layout P through LDS: C-layout -> A-fragment layout
#pragma unroll
        for (int r = 0; r < 8; ++r) {
            int row = r + g * 8;
            s_p[row * 32 + col]      = (_Float16)p0[r];
            s_p[row * 32 + 16 + col] = (_Float16)p1[r];
        }
        __syncthreads();
        v16h pa = frag_2x8(&s_p[col * 32 + g * 8], &s_p[col * 32 + 16 + g * 8]);

        // out = P(16x32) x Vsel(32x64): 4 n-tiles over E
#pragma unroll
        for (int t = 0; t < 4; ++t) {
            v8f z = {};
            v8f d = wmma16(pa, bv[t], z);
#pragma unroll
            for (int r = 0; r < 8; ++r) {
                int row = r + g * 8;
                if (tile * 16 + row < cnt) {
                    int tok2 = s_tok[row];
                    out[((size_t)(n * LL + tok2) * HH + h) * EE + t * 16 + col] = d[r];
                }
            }
        }
    }
}

// ---- launcher ---------------------------------------------------------------
extern "C" void kernel_launch(void* const* d_in, const int* in_sizes, int n_in,
                              void* d_out, int out_size, void* d_ws, size_t ws_size,
                              hipStream_t stream) {
    (void)in_sizes; (void)n_in; (void)out_size; (void)ws_size;
    const float* q = (const float*)d_in[0];
    const float* k = (const float*)d_in[1];
    const float* v = (const float*)d_in[2];
    const float* planes = (const float*)d_in[3];
    float* out = (float*)d_out;

    char* p = (char*)d_ws;
    auto alloc = [&](size_t bytes) -> char* {
        char* r = p; p += (bytes + 255) & ~(size_t)255; return r;
    };
    _Float16* qhh  = (_Float16*)alloc((size_t)NH * LL * EE * 2);
    _Float16* khh  = (_Float16*)alloc((size_t)NH * LL * EE * 2);
    _Float16* vhh  = (_Float16*)alloc((size_t)NH * LL * EE * 2);
    unsigned* bits = (unsigned*)alloc((size_t)NH * LL * 4);
    int* clusters  = (int*)alloc((size_t)NH * LL * 4);
    int* counts    = (int*)alloc((size_t)NH * CC * 4);
    float* qg      = (float*)alloc((size_t)NH * CC * EE * 4);
    _Float16* qgh  = (_Float16*)alloc((size_t)NH * CC * EE * 2);
    float* QKc     = (float*)alloc((size_t)NH * CC * LL * 4);
    int* topk      = (int*)alloc((size_t)NH * CC * TOPKK * 4);
    int* coff      = (int*)alloc((size_t)NH * CC * 4);
    int* cursor    = (int*)alloc((size_t)NH * CC * 4);
    int* tsorted   = (int*)alloc((size_t)NH * LL * 4);

    hipMemsetAsync(counts, 0, (size_t)NH * CC * 4, stream);
    hipMemsetAsync(qg, 0, (size_t)NH * CC * EE * 4, stream);

    k_bits<<<(NH * LL + 255) / 256, 256, 0, stream>>>(q, planes, bits);
    k_half<<<(NH * LL * EE + 255) / 256, 256, 0, stream>>>(q, k, v, qhh, khh, vhh);
    k_kmeans<<<NH, 256, 0, stream>>>(bits, clusters, counts);
    k_qg_acc<<<(NH * LL * EE + 255) / 256, 256, 0, stream>>>(q, clusters, qg);
    k_qg_norm<<<(NH * CC * EE + 255) / 256, 256, 0, stream>>>(qg, counts, qgh);
    k_qkc<<<dim3(LL / 64, CC / 16, NH), 32, 0, stream>>>(qgh, khh, QKc);
    k_topk<<<dim3(CC, NH), 256, 0, stream>>>(QKc, topk);
    k_scan<<<NH, 32, 0, stream>>>(counts, coff, cursor);
    k_scatter<<<(NH * LL + 255) / 256, 256, 0, stream>>>(clusters, cursor, tsorted);
    k_attn<<<dim3(CC, NH), 32, 0, stream>>>(qhh, khh, vhh, topk, counts, coff,
                                            tsorted, out);
}
